// GlobalFeatureGAT_23338852286623
// MI455X (gfx1250) — matrix operand order repaired and checked by reading, not compiled, not run on hardware
//
#include <hip/hip_runtime.h>
#include <hip/hip_bf16.h>

// ---------------- problem constants (match reference) ----------------
#define NN   30000
#define EE   480000
#define GG   64
#define LLEN 20
#define DD   32
#define SSC  64
#define HH1  4
#define CC   128
#define TOTF 224          // S + 5*D
#define HC   512          // H*C
#define ETOT (EE + NN)    // edges + self loops

typedef __attribute__((ext_vector_type(16))) __bf16 v16bf;
typedef __attribute__((ext_vector_type(8)))  float  v8f;
typedef int v4i __attribute__((vector_size(16)));   // matches builtin param type

// ---- async global->LDS path (guarded: falls back to sync copy) ----
#if defined(__has_builtin)
#if __has_builtin(__builtin_amdgcn_global_load_async_to_lds_b128) && \
    __has_builtin(__builtin_amdgcn_s_wait_asynccnt)
#define USE_ASYNC_LDS 1
#endif
#endif
#ifndef USE_ASYNC_LDS
#define USE_ASYNC_LDS 0
#endif

#if USE_ASYNC_LDS
typedef __attribute__((address_space(3))) v4i* lds_v4i_p;
#endif

// ---------------- helpers ----------------
__device__ __forceinline__ unsigned short f32_to_bf16_rne(float f) {
  unsigned int u = __float_as_uint(f);
  unsigned int r = u + 0x7FFFu + ((u >> 16) & 1u);
  return (unsigned short)(r >> 16);
}

__device__ __forceinline__ void atomic_max_f32(float* addr, float val) {
  unsigned int* ai = (unsigned int*)addr;
  unsigned int cur = *ai;
  while (__uint_as_float(cur) < val) {
    unsigned int old = atomicCAS(ai, cur, __float_as_uint(val));
    if (old == cur) break;
    cur = old;
  }
}

__device__ __forceinline__ void edge_nodes(const int* __restrict__ ei, int e,
                                           int& src, int& dst) {
  if (e < EE) { src = ei[e]; dst = ei[EE + e]; }
  else        { src = e - EE; dst = e - EE; }       // self loops appended
}

__device__ __forceinline__ float elu_f(float v) {
  return (v > 0.f) ? v : (__expf(v) - 1.f);
}

// ---------------- generic fill ----------------
__global__ void fill_f32_kernel(float* p, float v, long n) {
  long i = (long)blockIdx.x * blockDim.x + threadIdx.x;
  if (i < n) p[i] = v;
}

// ---------------- W[K,Nout] f32 -> Wt[Nout,K] bf16 ----------------
__global__ void transpose_to_bf16_kernel(const float* __restrict__ W,
                                         unsigned short* __restrict__ Wt,
                                         int K, int Nout) {
  int i = blockIdx.x * blockDim.x + threadIdx.x;
  if (i >= K * Nout) return;
  int nrow = i / K, k = i % K;
  Wt[i] = f32_to_bf16_rne(W[(size_t)k * Nout + nrow]);
}

// ---------------- embedding pools + concat + LayerNorm -> bf16 ----------------
__global__ __launch_bounds__(256)
void embed_ln_kernel(const float* __restrict__ xs,
                     const int* __restrict__ xop, const int* __restrict__ xsrc,
                     const int* __restrict__ xsink, const int* __restrict__ xstr,
                     const int* __restrict__ xpay,
                     const float* __restrict__ e_op, const float* __restrict__ e_src,
                     const float* __restrict__ e_sink, const float* __restrict__ e_str,
                     const float* __restrict__ e_pay,
                     const float* __restrict__ ln_scale, const float* __restrict__ ln_bias,
                     unsigned short* __restrict__ xb) {
  __shared__ float red[256];
  __shared__ float red2[256];
  __shared__ int   cnt[5];
  const int node = blockIdx.x;
  const int tid  = threadIdx.x;

  const int*   tabs[5] = {xop, xsrc, xsink, xstr, xpay};
  const float* embs[5] = {e_op, e_src, e_sink, e_str, e_pay};

  if (tid < 5) cnt[tid] = 0;
  __syncthreads();
  if (tid < 5 * LLEN) {
    int t = tid / LLEN, j = tid % LLEN;
    if (tabs[t][node * LLEN + j] != 0) atomicAdd(&cnt[t], 1);
  }
  __syncthreads();

  float f = 0.f;
  if (tid < SSC) {
    f = xs[(size_t)node * SSC + tid];
  } else if (tid < TOTF) {
    int t = (tid - SSC) / DD, c = (tid - SSC) % DD;
    float s = 0.f;
    for (int j = 0; j < LLEN; ++j) {
      int idx = tabs[t][node * LLEN + j];
      if (idx != 0) s += embs[t][idx * DD + c];
    }
    f = s / ((float)cnt[t] + 1e-9f);
  }
  red[tid]  = (tid < TOTF) ? f : 0.f;
  red2[tid] = (tid < TOTF) ? f * f : 0.f;
  __syncthreads();
  for (int s = 128; s > 0; s >>= 1) {
    if (tid < s) { red[tid] += red[tid + s]; red2[tid] += red2[tid + s]; }
    __syncthreads();
  }
  float mu  = red[0] / (float)TOTF;
  float var = red2[0] / (float)TOTF - mu * mu;
  float inv = rsqrtf(var + 1e-5f);
  if (tid < TOTF) {
    float y = (f - mu) * inv * ln_scale[tid] + ln_bias[tid];
    xb[(size_t)node * TOTF + tid] = f32_to_bf16_rne(y);
  }
}

// ---------------- WMMA GEMM: D[M,Nout] = A[M,K](bf16) * Bt[Nout,K]^T(bf16) ----
// 4 waves / block; each wave owns a 16x64 tile (4 f32 accumulators).
// The 64-row B panel (contiguous 64*K bf16 slab of Bt) is shared by all 4
// waves: staged once into LDS with GLOBAL_LOAD_ASYNC_TO_LDS_B128 (ASYNCcnt
// path), then inner-loop B fragments come from ds_load_b128 while A streams
// from global per-lane.
// A-fragment (ISA 7.12.2, 16-bit A 16x32): lane m=lane&15, half=lane>>4,
//   two 16B chunks at k+half*8 and k+16+half*8.
// B-fragment (column n per lane): two 16B chunks at k+half*8 and k+16+half*8.
__global__ __launch_bounds__(128)
void wmma_gemm_bf16_kernel(const unsigned short* __restrict__ A,
                           const unsigned short* __restrict__ Bt,
                           float* __restrict__ D,
                           int M, int K, int Nout) {
  __shared__ unsigned short lds_b[64 * HC];   // 64 KB max (K<=512)

  const int lane = threadIdx.x & 31;
  const int wave = threadIdx.x >> 5;
  const int mr   = lane & 15;
  const int half = lane >> 4;
  const int m0   = (blockIdx.x * 4 + wave) * 16;
  const int n0   = blockIdx.y * 64;

  // ---- cooperative B-panel staging: Bt[n0 .. n0+63][0..K) is contiguous ----
  {
    const unsigned short* gpanel = Bt + (size_t)n0 * K;
    const int chunks = (64 * K) / 8;            // 16B chunks
    for (int i = threadIdx.x; i < chunks; i += 128) {
#if USE_ASYNC_LDS
      __builtin_amdgcn_global_load_async_to_lds_b128(
          (v4i*)(gpanel + i * 8), (lds_v4i_p)(lds_b + i * 8), 0, 0);
#else
      *(uint4*)(lds_b + i * 8) = *(const uint4*)(gpanel + i * 8);
#endif
    }
#if USE_ASYNC_LDS
    __builtin_amdgcn_s_wait_asynccnt(0);
#endif
    __syncthreads();
  }

  if (m0 >= M) return;                          // wave-uniform; EXEC stays all-1s

  const unsigned short* arow  = A + (size_t)(m0 + mr) * K;
  const unsigned short* brow0 = lds_b + (size_t)( 0 + mr) * K;
  const unsigned short* brow1 = lds_b + (size_t)(16 + mr) * K;
  const unsigned short* brow2 = lds_b + (size_t)(32 + mr) * K;
  const unsigned short* brow3 = lds_b + (size_t)(48 + mr) * K;

  v8f acc0 = {}, acc1 = {}, acc2 = {}, acc3 = {};

  for (int k = 0; k < K; k += 32) {
    __builtin_prefetch(arow + k + 128, 0, 1);   // global_prefetch_b8

    v16bf av, bv0, bv1, bv2, bv3;
    {
      uint4 c0 = *(const uint4*)(arow + k + half * 8);
      uint4 c1 = *(const uint4*)(arow + k + 16 + half * 8);
      ((uint4*)&av)[0] = c0; ((uint4*)&av)[1] = c1;
    }
#define LOADB(dst, p)                                                     \
    { const uint4* q = (const uint4*)((p) + k + half * 8);                \
      ((uint4*)&(dst))[0] = q[0]; ((uint4*)&(dst))[1] = q[2]; }
    LOADB(bv0, brow0) LOADB(bv1, brow1) LOADB(bv2, brow2) LOADB(bv3, brow3)
#undef LOADB
    acc0 = __builtin_amdgcn_wmma_f32_16x16x32_bf16(false, av, false, bv0, (short)0, acc0, false, false);
    acc1 = __builtin_amdgcn_wmma_f32_16x16x32_bf16(false, av, false, bv1, (short)0, acc1, false, false);
    acc2 = __builtin_amdgcn_wmma_f32_16x16x32_bf16(false, av, false, bv2, (short)0, acc2, false, false);
    acc3 = __builtin_amdgcn_wmma_f32_16x16x32_bf16(false, av, false, bv3, (short)0, acc3, false, false);
  }

  // C/D layout: VGPR v -> row m0 + v + 8*half, column n0 + 16*t + mr
  for (int v = 0; v < 8; ++v) {
    float* drow = D + (size_t)(m0 + v + 8 * half) * Nout + n0 + mr;
    drow[0]  = acc0[v];
    drow[16] = acc1[v];
    drow[32] = acc2[v];
    drow[48] = acc3[v];
  }
}

// ------------- attention coefficients: one wave per (node, head) -------------
__global__ __launch_bounds__(128)
void attn_coef_kernel(const float* __restrict__ h,
                      const float* __restrict__ a_src,
                      const float* __restrict__ a_dst,
                      float* __restrict__ s, float* __restrict__ d,
                      int heads, int ld) {
  int wid  = blockIdx.x * (blockDim.x >> 5) + (threadIdx.x >> 5);
  int lane = threadIdx.x & 31;
  if (wid >= NN * heads) return;
  int n = wid / heads, hh = wid % heads;
  const float* hp = h + (size_t)n * ld + hh * CC;
  const float* as = a_src + hh * CC;
  const float* ad = a_dst + hh * CC;
  float ss = 0.f, dd = 0.f;
  for (int c = lane; c < CC; c += 32) {
    float v = hp[c];
    ss += v * as[c];
    dd += v * ad[c];
  }
  for (int o = 16; o > 0; o >>= 1) {
    ss += __shfl_xor(ss, o, 32);
    dd += __shfl_xor(dd, o, 32);
  }
  if (lane == 0) { s[wid] = ss; d[wid] = dd; }
}

// ---------------- edge pass 1: segment max of leaky_relu(s[src]+d[dst]) -----
__global__ void edge_max_kernel(const int* __restrict__ ei,
                                const float* __restrict__ s, const float* __restrict__ d,
                                float* __restrict__ m, int heads) {
  int i = blockIdx.x * blockDim.x + threadIdx.x;
  if (i >= ETOT * heads) return;
  int e = i / heads, hh = i % heads;
  int src, dst; edge_nodes(ei, e, src, dst);
  float v = s[src * heads + hh] + d[dst * heads + hh];
  v = (v > 0.f) ? v : 0.2f * v;
  atomic_max_f32(&m[dst * heads + hh], v);
}

// ---------------- edge pass 2: segment sum of exp(e - max) -------------------
__global__ void edge_den_kernel(const int* __restrict__ ei,
                                const float* __restrict__ s, const float* __restrict__ d,
                                const float* __restrict__ m, float* __restrict__ den,
                                int heads) {
  int i = blockIdx.x * blockDim.x + threadIdx.x;
  if (i >= ETOT * heads) return;
  int e = i / heads, hh = i % heads;
  int src, dst; edge_nodes(ei, e, src, dst);
  float v = s[src * heads + hh] + d[dst * heads + hh];
  v = (v > 0.f) ? v : 0.2f * v;
  atomicAdd(&den[dst * heads + hh], __expf(v - m[dst * heads + hh]));
}

// ---------------- edge pass 3: agg[dst] += alpha * h[src] --------------------
__global__ void edge_agg_kernel(const int* __restrict__ ei,
                                const float* __restrict__ s, const float* __restrict__ d,
                                const float* __restrict__ m, const float* __restrict__ den,
                                const float* __restrict__ h, float* __restrict__ agg,
                                int heads, int ld) {
  const int C4 = CC / 4;
  long total = (long)ETOT * heads * C4;
  long i = (long)blockIdx.x * blockDim.x + threadIdx.x;
  if (i >= total) return;
  int c4 = (int)(i % C4);
  long r  = i / C4;
  int hh  = (int)(r % heads);
  int e   = (int)(r / heads);
  int src, dst; edge_nodes(ei, e, src, dst);
  float v = s[src * heads + hh] + d[dst * heads + hh];
  v = (v > 0.f) ? v : 0.2f * v;
  float alpha = __expf(v - m[dst * heads + hh]) / den[dst * heads + hh];
  const float4 hv = *(const float4*)(h + (size_t)src * ld + hh * CC + c4 * 4);
  float* op = agg + (size_t)dst * ld + hh * CC + c4 * 4;
  atomicAdd(op + 0, alpha * hv.x);
  atomicAdd(op + 1, alpha * hv.y);
  atomicAdd(op + 2, alpha * hv.z);
  atomicAdd(op + 3, alpha * hv.w);
}

// ---------------- bias + ELU -> bf16 (layer-2 GEMM input) --------------------
__global__ void bias_elu_to_bf16_kernel(const float* __restrict__ agg,
                                        const float* __restrict__ bias,
                                        unsigned short* __restrict__ out, int F) {
  long i = (long)blockIdx.x * blockDim.x + threadIdx.x;
  if (i >= (long)NN * F) return;
  int f = (int)(i % F);
  out[i] = f32_to_bf16_rne(elu_f(agg[i] + bias[f]));
}

// ---------------- bias + ELU + global max pool -------------------------------
__global__ void pool_kernel(const float* __restrict__ agg2,
                            const float* __restrict__ b2,
                            const int* __restrict__ batch,
                            float* __restrict__ out) {
  int i = blockIdx.x * blockDim.x + threadIdx.x;
  if (i >= NN * CC) return;
  int n = i / CC, c = i % CC;
  float v = elu_f(agg2[i] + b2[c]);
  atomic_max_f32(&out[batch[n] * CC + c], v);
}

// ---------------- launcher ----------------
extern "C" void kernel_launch(void* const* d_in, const int* in_sizes, int n_in,
                              void* d_out, int out_size, void* d_ws, size_t ws_size,
                              hipStream_t stream) {
  const float* xs       = (const float*)d_in[0];
  const int*   xop      = (const int*)d_in[1];
  const int*   xsrc     = (const int*)d_in[2];
  const int*   xsink    = (const int*)d_in[3];
  const int*   xstr     = (const int*)d_in[4];
  const int*   xpay     = (const int*)d_in[5];
  const int*   ei       = (const int*)d_in[6];
  const int*   batch    = (const int*)d_in[7];
  const float* e_op     = (const float*)d_in[8];
  const float* e_src    = (const float*)d_in[9];
  const float* e_sink   = (const float*)d_in[10];
  const float* e_str    = (const float*)d_in[11];
  const float* e_pay    = (const float*)d_in[12];
  const float* ln_scale = (const float*)d_in[13];
  const float* ln_bias  = (const float*)d_in[14];
  const float* W1       = (const float*)d_in[15];
  const float* att_s1   = (const float*)d_in[16];
  const float* att_d1   = (const float*)d_in[17];
  const float* b1       = (const float*)d_in[18];
  const float* W2       = (const float*)d_in[19];
  const float* att_s2   = (const float*)d_in[20];
  const float* att_d2   = (const float*)d_in[21];
  const float* b2       = (const float*)d_in[22];
  float* out            = (float*)d_out;

  // workspace carve-out (256B aligned)
  char* ws = (char*)d_ws;
  size_t off = 0;
  auto carve = [&](size_t bytes) -> char* {
    char* p = ws + off;
    off = (off + bytes + 255) & ~(size_t)255;
    return p;
  };
  unsigned short* W1t  = (unsigned short*)carve((size_t)HC * TOTF * 2);
  unsigned short* W2t  = (unsigned short*)carve((size_t)CC * HC * 2);
  unsigned short* xb   = (unsigned short*)carve((size_t)NN * TOTF * 2);
  float* h1   = (float*)carve((size_t)NN * HC * 4);
  float* s1   = (float*)carve((size_t)NN * HH1 * 4);
  float* d1   = (float*)carve((size_t)NN * HH1 * 4);
  float* m1   = (float*)carve((size_t)NN * HH1 * 4);
  float* den1 = (float*)carve((size_t)NN * HH1 * 4);
  float* agg1 = (float*)carve((size_t)NN * HC * 4);
  unsigned short* x2b = (unsigned short*)carve((size_t)NN * HC * 2);
  float* h2   = (float*)carve((size_t)NN * CC * 4);
  float* s2   = (float*)carve((size_t)NN * 4);
  float* d2   = (float*)carve((size_t)NN * 4);
  float* m2   = (float*)carve((size_t)NN * 4);
  float* den2 = (float*)carve((size_t)NN * 4);
  float* agg2 = (float*)carve((size_t)NN * CC * 4);
  (void)ws_size;

  const int TB = 256;
  auto blocks = [](long n, int tb) { return (unsigned)((n + tb - 1) / tb); };

  // deterministic init every call (graph-replay safe)
  fill_f32_kernel<<<blocks((long)NN * HH1, TB), TB, 0, stream>>>(m1, -3.0e38f, (long)NN * HH1);
  fill_f32_kernel<<<blocks((long)NN * HH1, TB), TB, 0, stream>>>(den1, 0.f, (long)NN * HH1);
  fill_f32_kernel<<<blocks((long)NN * HC, TB), TB, 0, stream>>>(agg1, 0.f, (long)NN * HC);
  fill_f32_kernel<<<blocks((long)NN, TB), TB, 0, stream>>>(m2, -3.0e38f, (long)NN);
  fill_f32_kernel<<<blocks((long)NN, TB), TB, 0, stream>>>(den2, 0.f, (long)NN);
  fill_f32_kernel<<<blocks((long)NN * CC, TB), TB, 0, stream>>>(agg2, 0.f, (long)NN * CC);
  fill_f32_kernel<<<blocks((long)GG * CC, TB), TB, 0, stream>>>(out, -3.0e38f, (long)GG * CC);

  // weights -> transposed bf16
  transpose_to_bf16_kernel<<<blocks((long)TOTF * HC, TB), TB, 0, stream>>>(W1, W1t, TOTF, HC);
  transpose_to_bf16_kernel<<<blocks((long)HC * CC, TB), TB, 0, stream>>>(W2, W2t, HC, CC);

  // features: pools + layernorm -> bf16
  embed_ln_kernel<<<NN, 256, 0, stream>>>(xs, xop, xsrc, xsink, xstr, xpay,
                                          e_op, e_src, e_sink, e_str, e_pay,
                                          ln_scale, ln_bias, xb);

  // ---- layer 1 ----
  {
    dim3 grid((NN / 16 + 3) / 4, HC / 64);
    wmma_gemm_bf16_kernel<<<grid, 128, 0, stream>>>(xb, W1t, h1, NN, TOTF, HC);
  }
  attn_coef_kernel<<<blocks((long)NN * HH1 * 32, 128), 128, 0, stream>>>(
      h1, att_s1, att_d1, s1, d1, HH1, HC);
  edge_max_kernel<<<blocks((long)ETOT * HH1, TB), TB, 0, stream>>>(ei, s1, d1, m1, HH1);
  edge_den_kernel<<<blocks((long)ETOT * HH1, TB), TB, 0, stream>>>(ei, s1, d1, m1, den1, HH1);
  edge_agg_kernel<<<blocks((long)ETOT * HH1 * (CC / 4), TB), TB, 0, stream>>>(
      ei, s1, d1, m1, den1, h1, agg1, HH1, HC);
  bias_elu_to_bf16_kernel<<<blocks((long)NN * HC, TB), TB, 0, stream>>>(agg1, b1, x2b, HC);

  // ---- layer 2 ----
  {
    dim3 grid((NN / 16 + 3) / 4, CC / 64);
    wmma_gemm_bf16_kernel<<<grid, 128, 0, stream>>>(x2b, W2t, h2, NN, HC, CC);
  }
  attn_coef_kernel<<<blocks((long)NN * 32, 128), 128, 0, stream>>>(
      h2, att_s2, att_d2, s2, d2, 1, CC);
  edge_max_kernel<<<blocks((long)ETOT, TB), TB, 0, stream>>>(ei, s2, d2, m2, 1);
  edge_den_kernel<<<blocks((long)ETOT, TB), TB, 0, stream>>>(ei, s2, d2, m2, den2, 1);
  edge_agg_kernel<<<blocks((long)ETOT * (CC / 4), TB), TB, 0, stream>>>(
      ei, s2, d2, m2, den2, h2, agg2, 1, CC);

  // ---- bias + ELU + global max pool ----
  pool_kernel<<<blocks((long)NN * CC, TB), TB, 0, stream>>>(agg2, b2, batch, out);
}
